// BasicBlock_33200097198449
// MI455X (gfx1250) — matrix operand order repaired and checked by reading, not compiled, not run on hardware
//
#include <hip/hip_runtime.h>

typedef __bf16 bf16_t;
typedef __attribute__((ext_vector_type(16))) __bf16 v16bf;
typedef __attribute__((ext_vector_type(8)))  float  v8f;

#define NN 64
#define CC 256
#define HH 32
#define WW 32

union Frag16 { uint4 q[2]; v16bf v; };

// async global->LDS 16B per lane (ISA §15.18 op 98, ASYNCcnt-tracked).
// For generic pointers in the LDS aperture, addr[31:0] is the LDS address (ISA §10.2).
__device__ __forceinline__ void async_ld_b128(void* lds, const void* g) {
  unsigned loff = (unsigned)(unsigned long long)lds;
  asm volatile("global_load_async_to_lds_b128 %0, %1, off"
               :: "v"(loff), "v"(g) : "memory");
}
__device__ __forceinline__ void wait_async0() {
  asm volatile("s_wait_asynccnt 0x0" ::: "memory");
}

// ---------------- prep: pack ternary weights OIHW f32 -> [tap][cout][cin] bf16 ---
__global__ void prep_weights(const float* __restrict__ w1,
                             const float* __restrict__ w2,
                             bf16_t* __restrict__ w1p,
                             bf16_t* __restrict__ w2p) {
  int idx = blockIdx.x * blockDim.x + threadIdx.x;   // [0, 9*256*256)
  if (idx >= 9 * CC * CC) return;
  int t  = idx >> 16;
  int r  = idx & 0xFFFF;
  int co = r >> 8;
  int ci = r & 255;
  int src = co * (CC * 9) + ci * 9 + t;              // OIHW, t = kh*3+kw
  w1p[idx] = (bf16_t)w1[src];
  w2p[idx] = (bf16_t)w2[src];
}

// -------- prep: fold BN into per-channel scale/bias; also zero the zero-page -----
__global__ void prep_bn(const float* g1, const float* b1, const float* m1, const float* v1,
                        const float* g2, const float* b2, const float* m2, const float* v2,
                        float* s1, float* o1, float* s2, float* o2, float* zpage) {
  int c = threadIdx.x;
  float i1 = g1[c] * rsqrtf(v1[c] + 1e-5f);
  s1[c] = i1; o1[c] = b1[c] - m1[c] * i1;
  float i2 = g2[c] * rsqrtf(v2[c] + 1e-5f);
  s2[c] = i2; o2[c] = b2[c] - m2[c] * i2;
  zpage[c] = 0.f;                                    // 1 KB of zeros (covers 512 bf16)
}

// ---------------- prep: x NCHW f32 -> NHWC bf16 via LDS transpose -----------------
__global__ __launch_bounds__(256) void prep_x(const float* __restrict__ x,
                                              bf16_t* __restrict__ xb) {
  __shared__ bf16_t tile[CC][WW + 1];
  const int row = blockIdx.x;            // n*32 + h
  const int n = row >> 5, h = row & 31;
  const int tid = threadIdx.x;
  const int wl = tid & 31, cg = tid >> 5;
  const float* src = x + ((size_t)(n * CC) * HH + h) * WW;   // + c*1024 + w
  #pragma unroll 4
  for (int i = 0; i < 32; ++i) {
    int c = i * 8 + cg;
    tile[c][wl] = (bf16_t)src[(size_t)c * HH * WW + wl];
  }
  __syncthreads();
  bf16_t* dst = xb + (size_t)row * WW * CC;
  #pragma unroll
  for (int i = 0; i < 4; ++i) {
    int j = i * 256 + tid;
    int w = j >> 5;
    int cb = (j & 31) << 3;
    union P8 { bf16_t hv[8]; uint4 q; } p;
    #pragma unroll
    for (int r = 0; r < 8; ++r) p.hv[r] = tile[cb + r][w];
    *(uint4*)(dst + (size_t)w * CC + cb) = p.q;    // coalesced 16B stores
  }
}

// ---------------- main conv: implicit GEMM via v_wmma_f32_16x16x32_bf16 -----------
// Block = 8 waves, one cout tile (ct, 64 couts), 8 consecutive rows.
// Weights for the current tap live in LDS (32 KB), double-buffered across taps,
// staged with GLOBAL_LOAD_ASYNC_TO_LDS_B128: async transfers for tap t+1 issue
// before tap t's 64 WMMAs; s_wait_asynccnt 0 + one barrier per tap.
// LDS layout = exact WMMA fragment order: chunk idx = ((k*4+mt)*2+j)*32 + lane, 16B each.
// B (activations) via global with zero-page halo handling; K-loop double-buffered.
template <int MODE>
__global__ __launch_bounds__(256) void conv_bb(
    const bf16_t* __restrict__ act,     // NHWC bf16 input activations
    const bf16_t* __restrict__ wp,      // [9][256][256] bf16 weights
    const float*  __restrict__ scale,   // [256] BN scale
    const float*  __restrict__ bias,    // [256] BN bias
    const bf16_t* __restrict__ zpage,   // 512 zero bf16 values
    bf16_t*       __restrict__ hout,    // MODE 0 output
    const float*  __restrict__ xres,    // MODE 1 residual (NCHW f32)
    float*        __restrict__ out)     // MODE 1 output (NCHW f32)
{
  __shared__ __align__(16) bf16_t wbuf[2][16384];   // 2 x 32 KB tap buffers

  const int tid  = threadIdx.x;
  const int lane = tid & 31;
  const int lm = lane & 15, lh = lane >> 4;
  const int wave = tid >> 5;                        // 0..7
  const int ct  = blockIdx.x & 3;                   // cout tile of 64
  const int row = (blockIdx.x >> 2) * 8 + wave;     // n*32 + h, [0, 2048)
  const int n = row >> 5, h = row & 31;

  // staging mapping: thread -> (cout = tid&63, chunk quadrant = tid>>6)
  const int s_co = tid & 63;
  const int s_cq = tid >> 6;                        // 0..3
  const int s_mt = (tid >> 4) & 3;
  const int s_lm = tid & 15;
  const int s_j  = s_cq & 1;
  const int s_lh = s_cq >> 1;

  const v8f zero8 = {0.f, 0.f, 0.f, 0.f, 0.f, 0.f, 0.f, 0.f};
  v8f acc[4][2];
  #pragma unroll
  for (int mt = 0; mt < 4; ++mt)
    #pragma unroll
    for (int nt = 0; nt < 2; ++nt) acc[mt][nt] = zero8;

  const bf16_t* zb = zpage + lh * 16;   // per-lane zero-page base

  // prologue: async-stage tap 0 weights into buffer 0
  {
    const bf16_t* w0 = wp + (size_t)(ct * 64 + s_co) * CC + s_cq * 8;
    #pragma unroll
    for (int i = 0; i < 8; ++i) {
      int idx = ((i * 4 + s_mt) * 2 + s_j) * 32 + s_lh * 16 + s_lm;
      async_ld_b128(&wbuf[0][idx * 8], w0 + i * 32);
    }
    wait_async0();
  }
  __syncthreads();

  for (int t = 0; t < 9; ++t) {
    const int d = t & 1;
    const int dh = t / 3 - 1;
    const int dw = t - (t / 3) * 3 - 1;
    const int hh = h + dh;
    const bool rowok = (unsigned)hh < HH;

    // B pointers: invalid lanes read the zero page (no exec divergence)
    const bf16_t* bp[2];
    #pragma unroll
    for (int nt = 0; nt < 2; ++nt) {
      int wq = nt * 16 + lm + dw;
      bool ok = rowok && ((unsigned)wq < WW);
      int pos = (n * HH + (ok ? hh : 0)) * WW + (ok ? wq : 0);
      bp[nt] = ok ? (act + (size_t)pos * CC + lh * 16) : zb;
    }

    // async-stage tap t+1 weights into the inactive buffer (overlaps the WMMAs)
    if (t < 8) {
      const bf16_t* w0 = wp + (size_t)(t + 1) * CC * CC
                       + (size_t)(ct * 64 + s_co) * CC + s_cq * 8;
      #pragma unroll
      for (int i = 0; i < 8; ++i) {
        int idx = ((i * 4 + s_mt) * 2 + s_j) * 32 + s_lh * 16 + s_lm;
        async_ld_b128(&wbuf[d ^ 1][idx * 8], w0 + i * 32);
      }
    }

    Frag16 a[2][4], b[2][2];
    // chunk-0 prologue
    #pragma unroll
    for (int nt = 0; nt < 2; ++nt) {
      b[0][nt].q[0] = *(const uint4*)(bp[nt]);
      b[0][nt].q[1] = *(const uint4*)(bp[nt] + 8);
    }
    #pragma unroll
    for (int mt = 0; mt < 4; ++mt) {
      a[0][mt].q[0] = *(const uint4*)(&wbuf[d][(((0 * 4 + mt) * 2 + 0) * 32 + lane) * 8]);
      a[0][mt].q[1] = *(const uint4*)(&wbuf[d][(((0 * 4 + mt) * 2 + 1) * 32 + lane) * 8]);
    }

    #pragma unroll
    for (int k = 0; k < 8; ++k) {
      const int cur = k & 1, nx = cur ^ 1;
      if (k < 7) {                                  // prefetch next K-chunk
        const int c1 = (k + 1) * 32;
        #pragma unroll
        for (int nt = 0; nt < 2; ++nt) {
          b[nx][nt].q[0] = *(const uint4*)(bp[nt] + c1);
          b[nx][nt].q[1] = *(const uint4*)(bp[nt] + c1 + 8);
        }
        #pragma unroll
        for (int mt = 0; mt < 4; ++mt) {
          a[nx][mt].q[0] = *(const uint4*)(&wbuf[d][((((k + 1) * 4 + mt) * 2 + 0) * 32 + lane) * 8]);
          a[nx][mt].q[1] = *(const uint4*)(&wbuf[d][((((k + 1) * 4 + mt) * 2 + 1) * 32 + lane) * 8]);
        }
      }
      #pragma unroll
      for (int mt = 0; mt < 4; ++mt)
        #pragma unroll
        for (int nt = 0; nt < 2; ++nt)
          acc[mt][nt] = __builtin_amdgcn_wmma_f32_16x16x32_bf16(
              false, a[cur][mt].v, false, b[cur][nt].v, (short)0, acc[mt][nt],
              false, false);
    }

    // staged tap t+1 transfers must have landed before the block proceeds
    if (t < 8) wait_async0();
    __syncthreads();
  }

  // Epilogue. D layout: lane -> N = lm (position), VGPR r -> M = r + 8*lh (cout).
  #pragma unroll
  for (int mt = 0; mt < 4; ++mt) {
    const int cbase = ct * 64 + mt * 16 + lh * 8;   // 8 contiguous couts per lane
    float4 s0 = *(const float4*)(scale + cbase);
    float4 s1 = *(const float4*)(scale + cbase + 4);
    float4 o0 = *(const float4*)(bias + cbase);
    float4 o1 = *(const float4*)(bias + cbase + 4);
    float sc[8] = {s0.x, s0.y, s0.z, s0.w, s1.x, s1.y, s1.z, s1.w};
    float bs[8] = {o0.x, o0.y, o0.z, o0.w, o1.x, o1.y, o1.z, o1.w};
    #pragma unroll
    for (int nt = 0; nt < 2; ++nt) {
      const int w = nt * 16 + lm;
      if (MODE == 0) {
        union P8 { bf16_t hv[8]; uint4 q; } p;
        #pragma unroll
        for (int r = 0; r < 8; ++r) {
          float v = acc[mt][nt][r] * sc[r] + bs[r];
          v = v > 0.f ? v : 0.f;
          p.hv[r] = (bf16_t)v;
        }
        *(uint4*)(hout + ((size_t)row * WW + w) * CC + cbase) = p.q;  // NHWC bf16
      } else {
        #pragma unroll
        for (int r = 0; r < 8; ++r) {
          int c = cbase + r;
          size_t off = ((size_t)(n * CC + c) * HH + h) * WW + w;      // NCHW f32
          float v = acc[mt][nt][r] * sc[r] + bs[r] + xres[off];
          out[off] = v > 0.f ? v : 0.f;
        }
      }
    }
  }
}

// ---------------- workspace layout (bytes) ----------------------------------------
static const size_t OFF_W1P = 0;                       // 9*256*256 * 2 = 1,179,648
static const size_t OFF_W2P = 1179648;
static const size_t OFF_S1  = 2359296;                 // 256 * 4
static const size_t OFF_O1  = OFF_S1 + 1024;
static const size_t OFF_S2  = OFF_O1 + 1024;
static const size_t OFF_O2  = OFF_S2 + 1024;
static const size_t OFF_ZB  = OFF_O2 + 1024;           // 1 KB zero page
static const size_t OFF_XB  = OFF_ZB + 1024;           // 64*32*32*256 * 2 = 33,554,432
static const size_t OFF_HB  = OFF_XB + 33554432;       // same size

extern "C" void kernel_launch(void* const* d_in, const int* in_sizes, int n_in,
                              void* d_out, int out_size, void* d_ws, size_t ws_size,
                              hipStream_t stream) {
  const float* x   = (const float*)d_in[0];
  const float* w1  = (const float*)d_in[1];
  const float* g1  = (const float*)d_in[2];
  const float* b1  = (const float*)d_in[3];
  const float* rm1 = (const float*)d_in[4];
  const float* rv1 = (const float*)d_in[5];
  const float* w2  = (const float*)d_in[6];
  const float* g2  = (const float*)d_in[7];
  const float* b2  = (const float*)d_in[8];
  const float* rm2 = (const float*)d_in[9];
  const float* rv2 = (const float*)d_in[10];
  float* out = (float*)d_out;

  char* ws = (char*)d_ws;
  bf16_t* w1p = (bf16_t*)(ws + OFF_W1P);
  bf16_t* w2p = (bf16_t*)(ws + OFF_W2P);
  float*  s1  = (float*)(ws + OFF_S1);
  float*  o1  = (float*)(ws + OFF_O1);
  float*  s2  = (float*)(ws + OFF_S2);
  float*  o2  = (float*)(ws + OFF_O2);
  float*  zpf = (float*)(ws + OFF_ZB);
  bf16_t* zb  = (bf16_t*)(ws + OFF_ZB);
  bf16_t* xb  = (bf16_t*)(ws + OFF_XB);
  bf16_t* hb  = (bf16_t*)(ws + OFF_HB);

  // 1) pack weights to bf16 [tap][cout][cin]
  prep_weights<<<(9 * CC * CC + 255) / 256, 256, 0, stream>>>(w1, w2, w1p, w2p);
  // 2) fold BN + zero the zero-page
  prep_bn<<<1, 256, 0, stream>>>(g1, b1, rm1, rv1, g2, b2, rm2, rv2, s1, o1, s2, o2, zpf);
  // 3) x NCHW f32 -> NHWC bf16
  prep_x<<<NN * HH, 256, 0, stream>>>(x, xb);
  // 4) conv1 + bn1 + relu -> hb (bf16 NHWC)
  conv_bb<0><<<1024, 256, 0, stream>>>(xb, w1p, s1, o1, zb, hb, nullptr, nullptr);
  // 5) conv2 + bn2 + residual + relu -> out (f32 NCHW)
  conv_bb<1><<<1024, 256, 0, stream>>>(hb, w2p, s2, o2, zb, nullptr, x, out);
}
